// HierarchicalClassifier_15522011808339
// MI455X (gfx1250) — matrix operand order repaired and checked by reading, not compiled, not run on hardware
//
#include <hip/hip_runtime.h>
#include <hip/hip_bf16.h>
#include <math.h>

// ---------------------------------------------------------------------------
// Types for CDNA5 WMMA (wave32): v_wmma_f32_16x16x32_bf16
// ---------------------------------------------------------------------------
typedef __bf16 bf16_t;
typedef __attribute__((ext_vector_type(16))) __bf16 v16bf;
typedef __attribute__((ext_vector_type(8)))  __bf16 v8bf;
typedef __attribute__((ext_vector_type(8)))  float  v8f;

#define HDIM 128

// Guaranteed hardware f32 atomic-add (L2 atomic unit), no CAS-loop expansion.
// Non-returning form -> tracked by STOREcnt; s_endpgm implicitly waits idle.
__device__ __forceinline__ void atomic_add_f32_hw(float* p, float v) {
    asm volatile("global_atomic_add_f32 %0, %1, off"
                 :
                 : "v"(p), "v"(v)
                 : "memory");
}

// ---------------------------------------------------------------------------
// Degree / normalization
// ---------------------------------------------------------------------------
__global__ void deg_init_kernel(float* __restrict__ deg, int n) {
    int i = blockIdx.x * blockDim.x + threadIdx.x;
    if (i < n) deg[i] = 1.0f;   // self-loop contributes 1 to every node
}

__global__ void deg_accum_kernel(const long long* __restrict__ dst,
                                 float* __restrict__ deg, int nedges) {
    int e = blockIdx.x * blockDim.x + threadIdx.x;
    if (e < nedges) {
        int d = (int)dst[e];
        atomic_add_f32_hw(&deg[d], 1.0f);
    }
}

__global__ void dinv_kernel(const float* __restrict__ deg,
                            float* __restrict__ dinv, int n) {
    int i = blockIdx.x * blockDim.x + threadIdx.x;
    if (i < n) dinv[i] = rsqrtf(deg[i]);   // deg >= 1 always (self-loop)
}

// ---------------------------------------------------------------------------
// Elementwise converts
// ---------------------------------------------------------------------------
__global__ void f32_to_bf16_kernel(const float* __restrict__ x,
                                   bf16_t* __restrict__ y, size_t n) {
    size_t i = (size_t)blockIdx.x * blockDim.x + threadIdx.x;
    if (i < n) y[i] = (bf16_t)x[i];
}

__global__ void relu_to_bf16_kernel(const float* __restrict__ x,
                                    bf16_t* __restrict__ y, size_t n) {
    size_t i = (size_t)blockIdx.x * blockDim.x + threadIdx.x;
    if (i < n) y[i] = (bf16_t)fmaxf(x[i], 0.0f);
}

__global__ void relu_inplace_kernel(float* __restrict__ x, size_t n) {
    size_t i = (size_t)blockIdx.x * blockDim.x + threadIdx.x;
    if (i < n) x[i] = fmaxf(x[i], 0.0f);
}

// ---------------------------------------------------------------------------
// Pack W[128x128] (row-major, K x N) into WMMA B-fragment layout.
// Fragment f = nTile*4 + kStep holds 32 lanes x 16 bf16 contiguously:
//   lane l -> column n = nTile*16 + (l&15), K = kStep*32 + (l>>4)*8 + (j&7) + (j>>3)*16
// ---------------------------------------------------------------------------
__global__ void pack_w_kernel(const float* __restrict__ W,
                              bf16_t* __restrict__ Wp) {
    int t = blockIdx.x * blockDim.x + threadIdx.x;    // 0 .. 8*4*512-1
    if (t >= 8 * 4 * 512) return;
    int frag   = t >> 9;
    int within = t & 511;
    int lane   = within >> 4;
    int j      = within & 15;
    int nTile  = frag >> 2;
    int kStep  = frag & 3;
    int n = nTile * 16 + (lane & 15);
    int k = kStep * 32 + (lane >> 4) * 8 + (j & 7) + (j >> 3) * 16;
    Wp[t] = (bf16_t)W[k * HDIM + n];
}

// ---------------------------------------------------------------------------
// WMMA GEMM: C[nrows x 128] = A[nrows x 128](bf16, row-major) @ W(packed bf16)
// One wave per 16x16 tile; 8 waves/block span the full 128-wide output row.
// K = 128 in 4 steps of 32 via v_wmma_f32_16x16x32_bf16 (f32 accumulate).
// ---------------------------------------------------------------------------
__global__ __launch_bounds__(256)
void gemm_bf16_wmma_kernel(const bf16_t* __restrict__ A,
                           const bf16_t* __restrict__ Wp,
                           float* __restrict__ C) {
    const int lane  = threadIdx.x & 31;
    const int nTile = threadIdx.x >> 5;          // 0..7
    const int mBase = blockIdx.x * 16;
    const int half  = lane >> 4;

    // A fragment (ISA 16-bit A 16x32 layout): lane holds two contiguous
    // 8-element K chunks at K = k0 + half*8 and K = k0 + 16 + half*8.
    const int row = mBase + (lane & 15);
    const bf16_t* arow  = A  + (size_t)row * HDIM + half * 8;
    const bf16_t* bfrag = Wp + ((size_t)nTile * 4) * 512 + (size_t)lane * 16;

    v8f acc = {};
#pragma unroll
    for (int kk = 0; kk < 4; ++kk) {
        v8bf c0 = *(const v8bf*)(arow + kk * 32);
        v8bf c1 = *(const v8bf*)(arow + kk * 32 + 16);
        v16bf a;
#pragma unroll
        for (int i = 0; i < 8; ++i) { a[i] = c0[i]; a[8 + i] = c1[i]; }
        v16bf b = *(const v16bf*)(bfrag + kk * 512);
        acc = __builtin_amdgcn_wmma_f32_16x16x32_bf16(
            false, a, false, b, (short)0, acc, false, false);
    }

    // C/D layout: lane l, vgpr r -> (M = mBase + (l>>4)*8 + r, N = nTile*16 + (l&15))
    const int n  = nTile * 16 + (lane & 15);
    const int m0 = mBase + half * 8;
#pragma unroll
    for (int r = 0; r < 8; ++r)
        C[(size_t)(m0 + r) * HDIM + n] = acc[r];
}

// ---------------------------------------------------------------------------
// agg[i][h] = b[h] + dinv[i]^2 * xw[i][h]   (self-loop message + bias)
// ---------------------------------------------------------------------------
__global__ void init_agg_kernel(const float* __restrict__ xw,
                                const float* __restrict__ dinv,
                                const float* __restrict__ bias,
                                float* __restrict__ agg, size_t total) {
    size_t idx = (size_t)blockIdx.x * blockDim.x + threadIdx.x;
    if (idx >= total) return;
    size_t i = idx >> 7;
    int    h = (int)(idx & 127);
    float  di = dinv[i];
    agg[idx] = bias[h] + di * di * xw[idx];
}

// ---------------------------------------------------------------------------
// Edge aggregation: one wave per edge, lane handles 4 features.
// agg[dst] += dinv[src]*dinv[dst] * xw[src]  (global_atomic_add_f32 in L2)
// ---------------------------------------------------------------------------
__global__ __launch_bounds__(256)
void edge_agg_kernel(const long long* __restrict__ src,
                     const long long* __restrict__ dst,
                     const float* __restrict__ dinv,
                     const float* __restrict__ xw,
                     float* __restrict__ agg, int nedges) {
    const int lane = threadIdx.x & 31;
    const int e = blockIdx.x * (blockDim.x >> 5) + (threadIdx.x >> 5);
    if (e >= nedges) return;
    const int s = (int)src[e];
    const int d = (int)dst[e];
    const float norm = dinv[s] * dinv[d];
    const float4 v = *(const float4*)(xw + (size_t)s * HDIM + lane * 4);
    float* out = agg + (size_t)d * HDIM + lane * 4;
    atomic_add_f32_hw(out + 0, norm * v.x);
    atomic_add_f32_hw(out + 1, norm * v.y);
    atomic_add_f32_hw(out + 2, norm * v.z);
    atomic_add_f32_hw(out + 3, norm * v.w);
}

// ---------------------------------------------------------------------------
// Heads: det/und/root linears + log-softmax routing. One thread per node.
// ---------------------------------------------------------------------------
__global__ __launch_bounds__(256)
void heads_kernel(const float* __restrict__ h,
                  const float* __restrict__ Wdet, const float* __restrict__ bdet,
                  const float* __restrict__ Wund, const float* __restrict__ bund,
                  const float* __restrict__ Wroot, const float* __restrict__ broot,
                  float* __restrict__ out, int n) {
    __shared__ float sWdet[256], sWund[256], sWroot[512];
    for (int t = threadIdx.x; t < 256; t += 256) { sWdet[t] = Wdet[t]; sWund[t] = Wund[t]; }
    for (int t = threadIdx.x; t < 512; t += 256) sWroot[t] = Wroot[t];
    __syncthreads();

    int i = blockIdx.x * blockDim.x + threadIdx.x;
    if (i >= n) return;
    const float* hr = h + (size_t)i * HDIM;

    float d0 = bdet[0], d1 = bdet[1];
    float u0 = bund[0], u1 = bund[1];
    float r0 = broot[0], r1 = broot[1], r2 = broot[2], r3 = broot[3];
#pragma unroll 4
    for (int k = 0; k < HDIM; ++k) {
        float hv = hr[k];
        d0 = fmaf(hv, sWdet[k * 2 + 0], d0);
        d1 = fmaf(hv, sWdet[k * 2 + 1], d1);
        u0 = fmaf(hv, sWund[k * 2 + 0], u0);
        u1 = fmaf(hv, sWund[k * 2 + 1], u1);
        r0 = fmaf(hv, sWroot[k * 4 + 0], r0);
        r1 = fmaf(hv, sWroot[k * 4 + 1], r1);
        r2 = fmaf(hv, sWroot[k * 4 + 2], r2);
        r3 = fmaf(hv, sWroot[k * 4 + 3], r3);
    }

    // det log-softmax (2 classes)
    float m2  = fmaxf(d0, d1);
    float lse2 = m2 + logf(expf(d0 - m2) + expf(d1 - m2));
    out[(size_t)i * 2 + 0] = d0 - lse2;
    out[(size_t)i * 2 + 1] = d1 - lse2;

    // routing: argmax(det) (first index wins on ties -> mask iff d1 > d0)
    const bool mask = d1 > d0;
    const float NEGINF = -__builtin_inff();
    float c0 = mask ? NEGINF : u0;
    float c1 = mask ? r0 : u1;
    float c2 = mask ? r1 : NEGINF;
    float c3 = mask ? r2 : NEGINF;
    float c4 = mask ? r3 : NEGINF;
    float mm = fmaxf(fmaxf(fmaxf(c0, c1), fmaxf(c2, c3)), c4);
    float s = expf(c0 - mm) + expf(c1 - mm) + expf(c2 - mm) +
              expf(c3 - mm) + expf(c4 - mm);
    float lse5 = mm + logf(s);
    float* ro = out + (size_t)n * 2 + (size_t)i * 5;
    ro[0] = c0 - lse5;
    ro[1] = c1 - lse5;
    ro[2] = c2 - lse5;
    ro[3] = c3 - lse5;
    ro[4] = c4 - lse5;
}

// ---------------------------------------------------------------------------
// Launch
// ---------------------------------------------------------------------------
extern "C" void kernel_launch(void* const* d_in, const int* in_sizes, int n_in,
                              void* d_out, int out_size, void* d_ws, size_t ws_size,
                              hipStream_t stream) {
    const float*     x     = (const float*)d_in[0];
    const long long* eidx  = (const long long*)d_in[1];   // int64 per reference
    const float*     W1    = (const float*)d_in[2];
    const float*     b1    = (const float*)d_in[3];
    const float*     W2    = (const float*)d_in[4];
    const float*     b2    = (const float*)d_in[5];
    const float*     Wdet  = (const float*)d_in[6];
    const float*     bdet  = (const float*)d_in[7];
    const float*     Wund  = (const float*)d_in[8];
    const float*     bund  = (const float*)d_in[9];
    const float*     Wroot = (const float*)d_in[10];
    const float*     broot = (const float*)d_in[11];
    float* out = (float*)d_out;

    const int N = in_sizes[0] / HDIM;        // 50000
    const int E = in_sizes[1] / 2;           // 800000
    const long long* esrc = eidx;
    const long long* edst = eidx + E;

    // workspace carve-up (~64.5 MB)
    char* w = (char*)d_ws;
    float*  deg  = (float*)w;   w += (size_t)N * 4;
    float*  dinv = (float*)w;   w += (size_t)N * 4;
    bf16_t* Abf  = (bf16_t*)w;  w += (size_t)N * HDIM * 2;
    bf16_t* Wp1  = (bf16_t*)w;  w += 32 * 512 * 2;
    bf16_t* Wp2  = (bf16_t*)w;  w += 32 * 512 * 2;
    float*  xw   = (float*)w;   w += (size_t)N * HDIM * 4;
    float*  agg  = (float*)w;   w += (size_t)N * HDIM * 4;

    const size_t NH = (size_t)N * HDIM;
    const int TB = 256;
    const int gN   = (N + TB - 1) / TB;
    const int gE   = (E + TB - 1) / TB;
    const int gNH  = (int)((NH + TB - 1) / TB);
    const int gEdgeWave = (E + (TB / 32) - 1) / (TB / 32);  // 1 wave / edge
    const int gGemm = N / 16;                               // 3125 tile-rows

    // --- normalization ---
    deg_init_kernel<<<gN, TB, 0, stream>>>(deg, N);
    deg_accum_kernel<<<gE, TB, 0, stream>>>(edst, deg, E);
    dinv_kernel<<<gN, TB, 0, stream>>>(deg, dinv, N);

    // --- pack weights, convert input ---
    pack_w_kernel<<<(8 * 4 * 512 + TB - 1) / TB, TB, 0, stream>>>(W1, Wp1);
    pack_w_kernel<<<(8 * 4 * 512 + TB - 1) / TB, TB, 0, stream>>>(W2, Wp2);
    f32_to_bf16_kernel<<<gNH, TB, 0, stream>>>(x, Abf, NH);

    // --- layer 1 ---
    gemm_bf16_wmma_kernel<<<gGemm, TB, 0, stream>>>(Abf, Wp1, xw);
    init_agg_kernel<<<gNH, TB, 0, stream>>>(xw, dinv, b1, agg, NH);
    edge_agg_kernel<<<gEdgeWave, TB, 0, stream>>>(esrc, edst, dinv, xw, agg, E);
    relu_to_bf16_kernel<<<gNH, TB, 0, stream>>>(agg, Abf, NH);

    // --- layer 2 ---
    gemm_bf16_wmma_kernel<<<gGemm, TB, 0, stream>>>(Abf, Wp2, xw);
    init_agg_kernel<<<gNH, TB, 0, stream>>>(xw, dinv, b2, agg, NH);
    edge_agg_kernel<<<gEdgeWave, TB, 0, stream>>>(esrc, edst, dinv, xw, agg, E);
    relu_inplace_kernel<<<gNH, TB, 0, stream>>>(agg, NH);

    // --- heads + log-softmax routing ---
    heads_kernel<<<gN, TB, 0, stream>>>(agg, Wdet, bdet, Wund, bund,
                                        Wroot, broot, out, N);
}
